// MultiCrossAttentionBlock_83184926589352
// MI455X (gfx1250) — matrix operand order repaired and checked by reading, not compiled, not run on hardware
//
#include <hip/hip_runtime.h>
#include <hip/hip_bf16.h>
#include <math.h>

typedef __attribute__((ext_vector_type(16))) _Float16 v16h;
typedef __attribute__((ext_vector_type(8)))  float    v8f;

#define B_        16
#define N_L_      4097   // x0 rows per batch
#define N_KV      4098   // cls_m_proj + x0 rows
#define N_K1      1025   // cls_s_proj + p_m rows
#define DIM_S_    256
#define DIM_M_    512
#define DIM_L_    1024

// ---------------------------------------------------------------------------
// wave / block reductions (wave32)
// ---------------------------------------------------------------------------
__device__ __forceinline__ float waveRedSum(float v) {
#pragma unroll
  for (int o = 16; o > 0; o >>= 1) v += __shfl_xor(v, o, 32);
  return v;
}
__device__ __forceinline__ float waveRedMax(float v) {
#pragma unroll
  for (int o = 16; o > 0; o >>= 1) v = fmaxf(v, __shfl_xor(v, o, 32));
  return v;
}
__device__ float blockRedSum(float v, float* sh) {
  int lane = threadIdx.x & 31, w = threadIdx.x >> 5;
  int nw = (blockDim.x + 31) >> 5;
  v = waveRedSum(v);
  __syncthreads();
  if (lane == 0) sh[w] = v;
  __syncthreads();
  if (w == 0) {
    float r = (lane < nw) ? sh[lane] : 0.f;
    r = waveRedSum(r);
    if (lane == 0) sh[0] = r;
  }
  __syncthreads();
  float out = sh[0];
  __syncthreads();
  return out;
}
__device__ float blockRedMax(float v, float* sh) {
  int lane = threadIdx.x & 31, w = threadIdx.x >> 5;
  int nw = (blockDim.x + 31) >> 5;
  v = waveRedMax(v);
  __syncthreads();
  if (lane == 0) sh[w] = v;
  __syncthreads();
  if (w == 0) {
    float r = (lane < nw) ? sh[lane] : -INFINITY;
    r = waveRedMax(r);
    if (lane == 0) sh[0] = r;
  }
  __syncthreads();
  float out = sh[0];
  __syncthreads();
  return out;
}

// ---------------------------------------------------------------------------
// Generic 16-row GEMM using v_wmma_f32_16x16x32_f16 (one wave per 16x16 tile).
//   C(16 x N) = A(16 x K) * op(W)  [+ bias]
//   TRANSB=1: op(W)[k][n] = W[n*ldw + k]   (y = x @ W^T, weights are (out,in))
//   TRANSB=0: op(W)[k][n] = W[k*ldw + n]
// A,W are f32 in memory, converted to f16 fragments; f32 accumulate.
// Out-of-range columns of the last tile read the clamped row (N-1): the
// resulting garbage stays confined to C columns that are never stored, so all
// fragment loads are unconditional (no predicated loads in the hot loop).
// Fragment layouts used here (wave32; A/B use contiguous k, not the ISA table order):
//   A 16x32:  lane<16 holds row=lane, K {kk..kk+15};
//             lane>=16 holds K {kk+16..kk+31}
//   B 32x16:  lane holds column=lane&15, K {kk+16*(lane>>4) .. +15}
//   C 16x16:  lane holds column=lane&15, rows (lane<16 ? 0..7 : 8..15) in c[0..7]
// ---------------------------------------------------------------------------
template <int TRANSB>
__global__ void gemm16_wmma(const float* __restrict__ A, int lda,
                            const float* __restrict__ W, int ldw,
                            const float* __restrict__ bias,
                            float* __restrict__ C, int ldc, int K, int N) {
  const int lane  = threadIdx.x;      // 0..31, one wave per block
  const int row16 = lane & 15;
  const int hi    = lane >> 4;        // 0 or 1
  const int ncol  = blockIdx.x * 16 + row16;
  const bool colOK = (ncol < N);
  const int ncl   = colOK ? ncol : (N - 1);   // clamped for safe addressing

  v8f c = {0.f, 0.f, 0.f, 0.f, 0.f, 0.f, 0.f, 0.f};
  const int K32 = K & ~31;

#pragma unroll 2
  for (int kk = 0; kk < K32; kk += 32) {
    v16h a, bm;
    const float* ap = A + (size_t)row16 * lda + kk + hi * 16;
#pragma unroll
    for (int i = 0; i < 8; i++) {
      a[i]     = (_Float16)ap[i];
      a[8 + i] = (_Float16)ap[8 + i];
    }
    const int ks = kk + hi * 16;
    if (TRANSB) {
      const float* bp = W + (size_t)ncl * ldw + ks;
#pragma unroll
      for (int i = 0; i < 16; i++) bm[i] = (_Float16)bp[i];
    } else {
#pragma unroll
      for (int i = 0; i < 16; i++) bm[i] = (_Float16)W[(size_t)(ks + i) * ldw + ncl];
    }
    c = __builtin_amdgcn_wmma_f32_16x16x32_f16(false, a, false, bm,
                                               (short)0, c, false, false);
  }

  // K remainder (e.g. K = 1025 for the att1 projection): scalar rank-1 updates
  for (int k = K32; k < K; k++) {
    float bv = TRANSB ? W[(size_t)ncl * ldw + k] : W[(size_t)k * ldw + ncl];
#pragma unroll
    for (int j = 0; j < 8; j++) {
      int m = hi * 8 + j;
      c[j] += A[(size_t)m * lda + k] * bv;
    }
  }

  if (colOK) {
    float bb = bias ? bias[ncol] : 0.f;
#pragma unroll
    for (int j = 0; j < 8; j++) {
      int m = hi * 8 + j;
      C[(size_t)m * ldc + ncol] = c[j] + bb;
    }
  }
}

// ---------------------------------------------------------------------------
// logits1[b,k] = dot(row_k, qk1[b]) / 22 ;  row_0 = cls_s_proj[b], row_k = x1[b,k,:]
// one wave per (b,k), 512-float dot.
// ---------------------------------------------------------------------------
__global__ void logits1_kernel(const float* __restrict__ x1,
                               const float* __restrict__ cls_s_proj,
                               const float* __restrict__ qk1,
                               float* __restrict__ logits1) {
  const int b = blockIdx.y;
  const int wave = threadIdx.x >> 5, lane = threadIdx.x & 31;
  const int k = blockIdx.x * 8 + wave;
  if (k >= N_K1) return;
  const float* row = (k == 0) ? (cls_s_proj + (size_t)b * DIM_M_)
                              : (x1 + ((size_t)b * N_K1 + k) * DIM_M_);
  const float4* r4 = (const float4*)row;
  const float4* q4 = (const float4*)(qk1 + (size_t)b * DIM_M_);
  float acc = 0.f;
#pragma unroll
  for (int i = 0; i < 4; i++) {
    float4 rv = r4[lane + 32 * i];
    float4 qv = q4[lane + 32 * i];
    acc += rv.x * qv.x + rv.y * qv.y + rv.z * qv.z + rv.w * qv.w;
  }
  acc = waveRedSum(acc);
  if (lane == 0) logits1[(size_t)b * N_K1 + k] = acc * (1.0f / 22.0f);
}

// ---------------------------------------------------------------------------
// logits2[b,k] = dot(row_k, qk2[b]) / 32 ;  row_0 = cls_m_proj[b], row_k = x0[b,k-1,:]
// Streams x0 (268 MB) once. One wave per (b,k), 1024-float dot, b128 loads.
// ---------------------------------------------------------------------------
__global__ void logits2_kernel(const float* __restrict__ x0,
                               const float* __restrict__ cls_m_proj,
                               const float* __restrict__ qk2,
                               float* __restrict__ logits2) {
  const int b = blockIdx.y;
  const int wave = threadIdx.x >> 5, lane = threadIdx.x & 31;
  const int k = blockIdx.x * 8 + wave;
  if (k >= N_KV) return;
  const float* row = (k == 0) ? (cls_m_proj + (size_t)b * DIM_L_)
                              : (x0 + ((size_t)b * N_L_ + (k - 1)) * DIM_L_);
  const float4* r4 = (const float4*)row;
  const float4* q4 = (const float4*)(qk2 + (size_t)b * DIM_L_);
  float acc = 0.f;
#pragma unroll
  for (int i = 0; i < 8; i++) {
    float4 rv = r4[lane + 32 * i];
    float4 qv = q4[lane + 32 * i];
    acc += rv.x * qv.x + rv.y * qv.y + rv.z * qv.z + rv.w * qv.w;
  }
  acc = waveRedSum(acc);
  if (lane == 0) logits2[(size_t)b * N_KV + k] = acc * (1.0f / 32.0f);
}

// ---------------------------------------------------------------------------
// a1[b,:] = softmax(logits1[b,:])   (normalized, 1025 elems)
// ---------------------------------------------------------------------------
__global__ void softmax1_kernel(const float* __restrict__ logits1,
                                float* __restrict__ a1) {
  __shared__ float sh[33];
  const int b = blockIdx.x;
  const float* L = logits1 + (size_t)b * N_K1;
  float mx = -INFINITY;
  for (int k = threadIdx.x; k < N_K1; k += blockDim.x) mx = fmaxf(mx, L[k]);
  mx = blockRedMax(mx, sh);
  float s = 0.f;
  for (int k = threadIdx.x; k < N_K1; k += blockDim.x) s += __expf(L[k] - mx);
  s = blockRedSum(s, sh);
  float inv = 1.0f / s;
  for (int k = threadIdx.x; k < N_K1; k += blockDim.x)
    a1[(size_t)b * N_K1 + k] = __expf(L[k] - mx) * inv;
}

// ---------------------------------------------------------------------------
// wa[b,k] = 0.3*att1p[b,k] + 0.7*softmax(logits2[b,:])[k]    (4098 elems)
// ---------------------------------------------------------------------------
__global__ void softmax2_wa_kernel(const float* __restrict__ logits2,
                                   const float* __restrict__ att1p,
                                   float* __restrict__ wa) {
  __shared__ float sh[33];
  const int b = blockIdx.x;
  const float* L = logits2 + (size_t)b * N_KV;
  float mx = -INFINITY;
  for (int k = threadIdx.x; k < N_KV; k += blockDim.x) mx = fmaxf(mx, L[k]);
  mx = blockRedMax(mx, sh);
  float s = 0.f;
  for (int k = threadIdx.x; k < N_KV; k += blockDim.x) s += __expf(L[k] - mx);
  s = blockRedSum(s, sh);
  float inv = 0.7f / s;
  for (int k = threadIdx.x; k < N_KV; k += blockDim.x)
    wa[(size_t)b * N_KV + k] =
        0.3f * att1p[(size_t)b * N_KV + k] + __expf(L[k] - mx) * inv;
}

// ---------------------------------------------------------------------------
// Context partials: chunk c of 256 kv rows -> partial[(b*17+c)*1024 + d]
// Streams x0 (268 MB) once; each block reads whole 4KB rows coalesced (float4).
// ---------------------------------------------------------------------------
#define CTX_CHUNK 256
#define CTX_NCHUNK ((N_KV + CTX_CHUNK - 1) / CTX_CHUNK)   // 17

__global__ void ctx_partial_kernel(const float* __restrict__ x0,
                                   const float* __restrict__ cls_m_proj,
                                   const float* __restrict__ wa,
                                   float* __restrict__ partial) {
  __shared__ float wsh[CTX_CHUNK];
  const int b = blockIdx.y;
  const int k0 = blockIdx.x * CTX_CHUNK;
  const int t = threadIdx.x;               // 256 threads, thread t -> dims 4t..4t+3
  int kk = k0 + t;
  wsh[t] = (kk < N_KV) ? wa[(size_t)b * N_KV + kk] : 0.f;
  __syncthreads();
  float4 acc = {0.f, 0.f, 0.f, 0.f};
  const int kmax = min(CTX_CHUNK, N_KV - k0);
  for (int r = 0; r < kmax; r++) {
    int k = k0 + r;
    const float* row = (k == 0) ? (cls_m_proj + (size_t)b * DIM_L_)
                                : (x0 + ((size_t)b * N_L_ + (k - 1)) * DIM_L_);
    if (r + 2 < kmax) {   // prefetch two rows ahead (global_prefetch_b8)
      const float* nrow = x0 + ((size_t)b * N_L_ + (k0 + r + 1)) * DIM_L_;
      __builtin_prefetch(nrow + 4 * t, 0, 1);
    }
    float w = wsh[r];
    float4 v = ((const float4*)row)[t];
    acc.x += w * v.x; acc.y += w * v.y; acc.z += w * v.z; acc.w += w * v.w;
  }
  float4* out4 = (float4*)(partial + ((size_t)b * CTX_NCHUNK + blockIdx.x) * DIM_L_);
  out4[t] = acc;
}

__global__ void ctx_reduce_kernel(const float* __restrict__ partial,
                                  float* __restrict__ ctx) {
  const int b = blockIdx.x;
  const int t = threadIdx.x;               // 256 threads -> float4 each
  float4 acc = {0.f, 0.f, 0.f, 0.f};
  for (int c = 0; c < CTX_NCHUNK; c++) {
    float4 v = ((const float4*)(partial + ((size_t)b * CTX_NCHUNK + c) * DIM_L_))[t];
    acc.x += v.x; acc.y += v.y; acc.z += v.z; acc.w += v.w;
  }
  ((float4*)(ctx + (size_t)b * DIM_L_))[t] = acc;
}

// ---------------------------------------------------------------------------
// att3 softmax is over a length-1 axis => identically 1, so every output row
// equals g_s[b].  out[b, n, :] = g_s[b, :], n = 0..4096.  (67 MB write)
// grid = (chunks_per_batch, B); no 64-bit div/mod on the store path.
// ---------------------------------------------------------------------------
__global__ void broadcast_gs_kernel(const float4* __restrict__ gs4,
                                    float4* __restrict__ out4) {
  const int per_b = N_L_ * (DIM_S_ / 4);                 // 4097 * 64 = 262208
  const int b = blockIdx.y;
  int i = blockIdx.x * blockDim.x + threadIdx.x;
  if (i >= per_b) return;
  int d4 = i & (DIM_S_ / 4 - 1);                         // i % 64 (rows contiguous)
  out4[(size_t)b * per_b + i] = gs4[b * (DIM_S_ / 4) + d4];
}

// ---------------------------------------------------------------------------
extern "C" void kernel_launch(void* const* d_in, const int* in_sizes, int n_in,
                              void* d_out, int out_size, void* d_ws, size_t ws_size,
                              hipStream_t stream) {
  const float* x0     = (const float*)d_in[0];
  const float* x1     = (const float*)d_in[1];
  const float* x2     = (const float*)d_in[2];
  const float* f_s_w  = (const float*)d_in[3];
  const float* f_s_b  = (const float*)d_in[4];
  const float* f_m_w  = (const float*)d_in[5];
  const float* f_m_b  = (const float*)d_in[6];
  const float* Wq1    = (const float*)d_in[7];
  const float* Wk1    = (const float*)d_in[8];
  const float* Wq2    = (const float*)d_in[9];
  const float* Wk2    = (const float*)d_in[10];
  const float* Wv     = (const float*)d_in[11];
  const float* proj_w = (const float*)d_in[12];
  const float* proj_b = (const float*)d_in[13];
  const float* gs_w   = (const float*)d_in[14];
  const float* gs_b   = (const float*)d_in[15];

  float* ws = (float*)d_ws;
  float* w_cls_s = ws;             // 16*512
  float* w_q1    = ws + 8192;      // 16*512
  float* w_qk1   = ws + 16384;     // 16*512
  float* w_cls_m = ws + 24576;     // 16*1024
  float* w_q2    = ws + 40960;     // 16*1024
  float* w_qk2   = ws + 57344;     // 16*1024
  float* w_l1    = ws + 73728;     // 16*1025
  float* w_a1    = ws + 90128;     // 16*1025
  float* w_att1p = ws + 106528;    // 16*4098
  float* w_l2    = ws + 172096;    // 16*4098
  float* w_wa    = ws + 237664;    // 16*4098
  float* w_ctx   = ws + 303232;    // 16*1024
  float* w_outv  = ws + 319616;    // 16*1024
  float* w_gs    = ws + 336000;    // 16*256
  float* w_part  = ws + 340096;    // 17*16*1024

  const dim3 wb(32);

  // small-GEMM chain (M = 16 batches), all via WMMA f32<-f16
  gemm16_wmma<1><<<dim3(DIM_M_/16), wb, 0, stream>>>(x2, N_L_*DIM_S_, f_s_w, DIM_S_, f_s_b, w_cls_s, DIM_M_, DIM_S_, DIM_M_);
  gemm16_wmma<1><<<dim3(DIM_M_/16), wb, 0, stream>>>(w_cls_s, DIM_M_, Wq1, DIM_M_, nullptr, w_q1,  DIM_M_, DIM_M_, DIM_M_);
  gemm16_wmma<0><<<dim3(DIM_M_/16), wb, 0, stream>>>(w_q1,    DIM_M_, Wk1, DIM_M_, nullptr, w_qk1, DIM_M_, DIM_M_, DIM_M_);
  gemm16_wmma<1><<<dim3(DIM_L_/16), wb, 0, stream>>>(x1, N_K1*DIM_M_, f_m_w, DIM_M_, f_m_b, w_cls_m, DIM_L_, DIM_M_, DIM_L_);
  gemm16_wmma<1><<<dim3(DIM_L_/16), wb, 0, stream>>>(w_cls_m, DIM_L_, Wq2, DIM_L_, nullptr, w_q2,  DIM_L_, DIM_L_, DIM_L_);
  gemm16_wmma<0><<<dim3(DIM_L_/16), wb, 0, stream>>>(w_q2,    DIM_L_, Wk2, DIM_L_, nullptr, w_qk2, DIM_L_, DIM_L_, DIM_L_);

  // attention 1 (folded): logits, softmax, projection GEMM (K=1025, N=4098)
  logits1_kernel<<<dim3((N_K1 + 7)/8, B_), dim3(256), 0, stream>>>(x1, w_cls_s, w_qk1, w_l1);
  softmax1_kernel<<<dim3(B_), dim3(256), 0, stream>>>(w_l1, w_a1);
  gemm16_wmma<1><<<dim3((N_KV + 15)/16), wb, 0, stream>>>(w_a1, N_K1, proj_w, N_K1, proj_b, w_att1p, N_KV, N_K1, N_KV);

  // attention 2 (folded): stream x0 for logits, softmax+mix, stream x0 for ctx
  logits2_kernel<<<dim3((N_KV + 7)/8, B_), dim3(256), 0, stream>>>(x0, w_cls_m, w_qk2, w_l2);
  softmax2_wa_kernel<<<dim3(B_), dim3(1024), 0, stream>>>(w_l2, w_att1p, w_wa);
  ctx_partial_kernel<<<dim3(CTX_NCHUNK, B_), dim3(256), 0, stream>>>(x0, w_cls_m, w_wa, w_part);
  ctx_reduce_kernel<<<dim3(B_), dim3(256), 0, stream>>>(w_part, w_ctx);

  // out = ctx @ Wv^T ; g_s = out @ gs_w^T + gs_b   (WMMA)
  gemm16_wmma<1><<<dim3(DIM_L_/16), wb, 0, stream>>>(w_ctx,  DIM_L_, Wv,   DIM_L_, nullptr, w_outv, DIM_L_, DIM_L_, DIM_L_);
  gemm16_wmma<1><<<dim3(DIM_S_/16), wb, 0, stream>>>(w_outv, DIM_L_, gs_w, DIM_L_, gs_b,    w_gs,   DIM_S_, DIM_L_, DIM_S_);

  // att3 collapses to 1.0 -> broadcast g_s to all 4097 rows
  const int per_b4 = N_L_ * (DIM_S_ / 4);
  broadcast_gs_kernel<<<dim3((per_b4 + 255) / 256, B_), dim3(256), 0, stream>>>(
      (const float4*)w_gs, (float4*)d_out);
}
